// Bottleneck_DCN_MPCA_42107859370475
// MI455X (gfx1250) — matrix operand order repaired and checked by reading, not compiled, not run on hardware
//
#include <hip/hip_runtime.h>
#include <hip/hip_bf16.h>
#include <math.h>

typedef __attribute__((ext_vector_type(16))) __bf16 v16bf;
typedef __attribute__((ext_vector_type(8)))  float  v8f;

__device__ __forceinline__ float sigm(float x)   { return 1.f / (1.f + __expf(-x)); }
__device__ __forceinline__ float silu_f(float x) { return x * sigm(x); }

// ---------------------------------------------------------------------------
// Weight conversion: fp32 (rows, K) -> bf16 (rowsPad, K), zero padded rows.
// ---------------------------------------------------------------------------
__global__ void cvt_w_bf16(const float* __restrict__ w, __bf16* __restrict__ out,
                           int rows, int K, int rowsPad) {
    int i = blockIdx.x * blockDim.x + threadIdx.x;
    int total = rowsPad * K;
    if (i >= total) return;
    int r = i / K;
    out[i] = (r < rows) ? (__bf16)w[i] : (__bf16)0.0f;
}

// ---------------------------------------------------------------------------
// Offset conv: plain 3x3 pad-1 conv (Cin -> 27, padded to 32) as implicit GEMM.
// One wave per 16-pixel tile; A = pixels x K (K = c*9+k), B = K x oc (bf16
// pre-converted), D accumulated in fp32 via v_wmma_f32_16x16x32_bf16.
// ---------------------------------------------------------------------------
template <int CIN>
__global__ void __launch_bounds__(32)
offconv_wmma(const float* __restrict__ x,     // (B, CIN, 6400)
             const __bf16* __restrict__ wbf,  // (32, CIN*9) zero-padded bf16
             const float* __restrict__ bias,  // (27)
             float* __restrict__ om) {        // (B, 27, 6400)
    const int H = 80, W = 80, HW = 6400, K = CIN * 9;
    const int b    = blockIdx.y;
    const int pix0 = blockIdx.x * 16;
    const int l    = threadIdx.x;
    const int prow = l & 15, hi = l >> 4;
    const int p  = pix0 + prow;
    const int ph = p / W, pw = p % W;
    const float* xb = x + (size_t)b * CIN * HW;

    v8f acc0, acc1;
#pragma unroll
    for (int j = 0; j < 8; ++j) { acc0[j] = 0.f; acc1[j] = 0.f; }

    for (int K0 = 0; K0 < K; K0 += 32) {
        v16bf a;
#pragma unroll
        for (int e = 0; e < 16; ++e) {
            int Kidx = K0 + hi * 16 + e;
            int c = Kidx / 9, k = Kidx % 9;
            int yy = ph + (k / 3) - 1;
            int xx = pw + (k % 3) - 1;
            float v = 0.f;
            if (yy >= 0 && yy < H && xx >= 0 && xx < W) v = xb[c * HW + yy * W + xx];
            a[e] = (__bf16)v;
        }
        const __bf16* bp0 = wbf + (size_t)(0  + (l & 15)) * K + K0 + hi * 16;
        const __bf16* bp1 = wbf + (size_t)(16 + (l & 15)) * K + K0 + hi * 16;
        v16bf b0 = *(const v16bf*)bp0;
        v16bf b1 = *(const v16bf*)bp1;
        acc0 = __builtin_amdgcn_wmma_f32_16x16x32_bf16(false, a, false, b0, (short)0, acc0, false, false);
        acc1 = __builtin_amdgcn_wmma_f32_16x16x32_bf16(false, a, false, b1, (short)0, acc1, false, false);
    }
    const int n = l & 15;
#pragma unroll
    for (int t = 0; t < 2; ++t) {
        int oc = t * 16 + n;
        if (oc >= 27) continue;
        v8f acc = t ? acc1 : acc0;
        float bs = bias[oc];
#pragma unroll
        for (int r = 0; r < 8; ++r) {
            int pp = pix0 + hi * 8 + r;
            om[((size_t)b * 27 + oc) * HW + pp] = acc[r] + bs;
        }
    }
}

// ---------------------------------------------------------------------------
// MPCA stage 1: row/col/global means per (b, c) map.
// ---------------------------------------------------------------------------
__global__ void mpca_pool(const float* __restrict__ om, float* __restrict__ ph,
                          float* __restrict__ pw, float* __restrict__ pch) {
    int bc = blockIdx.x;          // b*27 + c
    int t  = threadIdx.x;         // 80 threads
    const float* base = om + (size_t)bc * 6400;
    float rs = 0.f, cs = 0.f;
    for (int w = 0; w < 80; ++w) rs += base[t * 80 + w];
    for (int h = 0; h < 80; ++h) cs += base[h * 80 + t];
    ph[bc * 80 + t] = rs * (1.f / 80.f);
    pw[bc * 80 + t] = cs * (1.f / 80.f);
    __shared__ float red[80];
    red[t] = rs;
    __syncthreads();
    if (t == 0) {
        float s = 0.f;
        for (int i = 0; i < 80; ++i) s += red[i];
        pch[bc] = s * (1.f / 6400.f);
    }
}

struct BN { const float *g, *b, *m, *v; };
struct MpcaP {
    const float* gw;  BN gbn;   // gap 1x1 (27x27)
    const float* hww; BN hbn;   // hw 3x1  (27x27x3)
    const float* pww; BN pbn;   // pool_hw 1x1 (27x27)
};

__device__ __forceinline__ float bn_apply(float x, const BN& bn, int c) {
    float s = bn.g[c] * rsqrtf(bn.v[c] + 1e-5f);
    return x * s + (bn.b[c] - bn.m[c] * s);
}

// ---------------------------------------------------------------------------
// MPCA stage 2: the whole 27x160 attention chain per batch, staged in LDS.
// Outputs per-(b,c) factors: sh[h], sw[w], sc (already sigmoided).
// ---------------------------------------------------------------------------
__global__ void mpca_mid(const float* __restrict__ ph, const float* __restrict__ pw,
                         const float* __restrict__ pch, MpcaP P,
                         float* __restrict__ sh, float* __restrict__ sw,
                         float* __restrict__ sc) {
    const int b   = blockIdx.x;
    const int tid = threadIdx.x;  // 256
    __shared__ float phw [27 * 160];
    __shared__ float phw2[27 * 160];
    __shared__ float wgt [27 * 160];
    __shared__ float pch2[27];

    for (int i = tid; i < 27 * 160; i += 256) {
        int c = i / 160, j = i % 160;
        phw[i] = (j < 80) ? ph[(b * 27 + c) * 80 + j] : pw[(b * 27 + c) * 80 + (j - 80)];
    }
    if (tid < 27) {
        float acc = 0.f;
        for (int ci = 0; ci < 27; ++ci) acc += P.gw[tid * 27 + ci] * pch[b * 27 + ci];
        pch2[tid] = silu_f(bn_apply(acc, P.gbn, tid));
    }
    __syncthreads();
    // hw conv 3x1, pad (1,0) along the 160-length concatenated axis
    for (int i = tid; i < 27 * 160; i += 256) {
        int co = i / 160, j = i % 160;
        float acc = 0.f;
        for (int ci = 0; ci < 27; ++ci) {
            const float* wr = P.hww + (co * 27 + ci) * 3;
            float v0 = (j > 0)   ? phw[ci * 160 + j - 1] : 0.f;
            float v1 =             phw[ci * 160 + j];
            float v2 = (j < 159) ? phw[ci * 160 + j + 1] : 0.f;
            acc = fmaf(wr[0], v0, fmaf(wr[1], v1, fmaf(wr[2], v2, acc)));
        }
        phw2[i] = silu_f(bn_apply(acc, P.hbn, co));
    }
    __syncthreads();
    // pool_hw 1x1 + sigmoid -> wgt
    for (int i = tid; i < 27 * 160; i += 256) {
        int co = i / 160, j = i % 160;
        float acc = 0.f;
        for (int ci = 0; ci < 27; ++ci) acc = fmaf(P.pww[co * 27 + ci], phw2[ci * 160 + j], acc);
        wgt[i] = sigm(silu_f(bn_apply(acc, P.pbn, co)));
    }
    __syncthreads();
    if (tid < 27) {
        float s = 0.f;
        for (int j = 0; j < 160; ++j) s += wgt[tid * 160 + j];
        sc[b * 27 + tid] = sigm(pch2[tid] * (s * (1.f / 160.f)));
    }
    for (int i = tid; i < 27 * 160; i += 256) {
        int c = i / 160, j = i % 160;
        float f = sigm(phw2[i] * wgt[i]);
        if (j < 80) sh[(b * 27 + c) * 80 + j]        = f;
        else        sw[(b * 27 + c) * 80 + (j - 80)] = f;
    }
}

// ---------------------------------------------------------------------------
// MPCA stage 3: apply attention factors elementwise to the offset map.
// ---------------------------------------------------------------------------
__global__ void mpca_apply(const float* __restrict__ om, const float* __restrict__ sh,
                           const float* __restrict__ sw, const float* __restrict__ sc,
                           float* __restrict__ om2, int total) {
    int i = blockIdx.x * blockDim.x + threadIdx.x;
    if (i >= total) return;
    int p = i % 6400, bc = i / 6400;
    int h = p / 80, w = p % 80;
    om2[i] = om[i] * sh[bc * 80 + h] * sw[bc * 80 + w] * sc[bc];
}

// ---------------------------------------------------------------------------
// Deformable conv as fused implicit GEMM.
// One wave per 16-pixel tile computes all OCT 16-wide output-channel tiles,
// sharing a single gathered A fragment. The (k, pixel) bilinear corner
// indices / mask-scaled weights are precomputed once in LDS and reused
// across all Cin. Epilogue fuses bias + BN + SiLU (+ residual).
// ---------------------------------------------------------------------------
template <int CIN, int OCT, bool RESID>
__global__ void __launch_bounds__(32)
deform_wmma(const float* __restrict__ x,      // (B, CIN, 6400)
            const float* __restrict__ om,     // (B, 27, 6400) attention-modulated
            const __bf16* __restrict__ wbf,   // (OCT*16, CIN*9) bf16
            const float* __restrict__ bias,
            const float* __restrict__ bng, const float* __restrict__ bnb,
            const float* __restrict__ bnm, const float* __restrict__ bnv,
            const float* __restrict__ resid,  // residual (or nullptr)
            float* __restrict__ out) {        // (B, OCT*16, 6400)
    const int H = 80, W = 80, HW = 6400, K = CIN * 9, COUT = OCT * 16;
    const int b    = blockIdx.y;
    const int pix0 = blockIdx.x * 16;
    const int l    = threadIdx.x;
    const int prow = l & 15, hi = l >> 4;

    __shared__ int   sIdx[144][4];
    __shared__ float sWgt[144][4];

    const float* omb = om + (size_t)b * 27 * HW;
    for (int e = l; e < 144; e += 32) {
        int k = e / 16, pr = e % 16;
        int p  = pix0 + pr;
        int hh = p / W, ww = p % W;
        float dy = omb[(2 * k)     * HW + p];
        float dx = omb[(2 * k + 1) * HW + p];
        float m  = sigm(omb[(18 + k) * HW + p]);
        float py = dy + (float)(hh + k / 3 - 1);
        float px = dx + (float)(ww + k % 3 - 1);
        float y0f = floorf(py), x0f = floorf(px);
        int   y0 = (int)y0f,   x0 = (int)x0f;
        float wy = py - y0f,   wx = px - x0f;
        float wv[4] = { (1.f - wy) * (1.f - wx) * m, (1.f - wy) * wx * m,
                        wy * (1.f - wx) * m,         wy * wx * m };
#pragma unroll
        for (int j = 0; j < 4; ++j) {
            int yy = y0 + (j >> 1), xx = x0 + (j & 1);
            bool ok = (yy >= 0 && yy < H && xx >= 0 && xx < W);
            int yc = min(max(yy, 0), H - 1), xc = min(max(xx, 0), W - 1);
            sIdx[e][j] = yc * W + xc;
            sWgt[e][j] = ok ? wv[j] : 0.f;
        }
    }
    __syncthreads();

    const float* xb = x + (size_t)b * CIN * HW;
    v8f acc[OCT];
#pragma unroll
    for (int t = 0; t < OCT; ++t)
#pragma unroll
        for (int j = 0; j < 8; ++j) acc[t][j] = 0.f;

    for (int K0 = 0; K0 < K; K0 += 32) {
        v16bf a;
#pragma unroll
        for (int e = 0; e < 16; ++e) {
            int Kidx = K0 + hi * 16 + e;
            int c = Kidx / 9, k = Kidx % 9;
            int s = k * 16 + prow;
            const float* base = xb + c * HW;
            float v = sWgt[s][0] * base[sIdx[s][0]] + sWgt[s][1] * base[sIdx[s][1]]
                    + sWgt[s][2] * base[sIdx[s][2]] + sWgt[s][3] * base[sIdx[s][3]];
            a[e] = (__bf16)v;
        }
#pragma unroll
        for (int t = 0; t < OCT; ++t) {
            const __bf16* bp = wbf + (size_t)(t * 16 + (l & 15)) * K + K0 + hi * 16;
            v16bf bt = *(const v16bf*)bp;
            acc[t] = __builtin_amdgcn_wmma_f32_16x16x32_bf16(false, a, false, bt, (short)0, acc[t], false, false);
        }
    }

    const int n = l & 15;
#pragma unroll
    for (int t = 0; t < OCT; ++t) {
        int oc = t * 16 + n;
        float scale = bng[oc] * rsqrtf(bnv[oc] + 1e-5f);
        float shift = bnb[oc] - bnm[oc] * scale;
        float bs = bias[oc];
#pragma unroll
        for (int r = 0; r < 8; ++r) {
            int pp = pix0 + hi * 8 + r;
            float y = (acc[t][r] + bs) * scale + shift;
            y = y * sigm(y);
            if (RESID) y += resid[((size_t)b * COUT + oc) * HW + pp];
            out[((size_t)b * COUT + oc) * HW + pp] = y;
        }
    }
}

// ---------------------------------------------------------------------------
// Orchestration
// ---------------------------------------------------------------------------
extern "C" void kernel_launch(void* const* d_in, const int* in_sizes, int n_in,
                              void* d_out, int out_size, void* d_ws, size_t ws_size,
                              hipStream_t stream) {
    (void)in_sizes; (void)n_in; (void)out_size; (void)ws_size;
    const float* X = (const float*)d_in[0];
    const int B = 2, HW = 6400;

    // --- input pointer map (depth-first dict order of setup_inputs) ---
    const float* W1    = (const float*)d_in[1];
    const float* B1    = (const float*)d_in[2];
    const float* OFFW1 = (const float*)d_in[3];
    const float* OFFB1 = (const float*)d_in[4];
    MpcaP M1 = { (const float*)d_in[5],
                 { (const float*)d_in[6],  (const float*)d_in[7],  (const float*)d_in[8],  (const float*)d_in[9]  },
                 (const float*)d_in[10],
                 { (const float*)d_in[11], (const float*)d_in[12], (const float*)d_in[13], (const float*)d_in[14] },
                 (const float*)d_in[15],
                 { (const float*)d_in[16], (const float*)d_in[17], (const float*)d_in[18], (const float*)d_in[19] } };
    const float* BN1g = (const float*)d_in[20]; const float* BN1b = (const float*)d_in[21];
    const float* BN1m = (const float*)d_in[22]; const float* BN1v = (const float*)d_in[23];
    const float* W2    = (const float*)d_in[24];
    const float* B2    = (const float*)d_in[25];
    const float* OFFW2 = (const float*)d_in[26];
    const float* OFFB2 = (const float*)d_in[27];
    MpcaP M2 = { (const float*)d_in[28],
                 { (const float*)d_in[29], (const float*)d_in[30], (const float*)d_in[31], (const float*)d_in[32] },
                 (const float*)d_in[33],
                 { (const float*)d_in[34], (const float*)d_in[35], (const float*)d_in[36], (const float*)d_in[37] },
                 (const float*)d_in[38],
                 { (const float*)d_in[39], (const float*)d_in[40], (const float*)d_in[41], (const float*)d_in[42] } };
    const float* BN2g = (const float*)d_in[43]; const float* BN2b = (const float*)d_in[44];
    const float* BN2m = (const float*)d_in[45]; const float* BN2v = (const float*)d_in[46];

    // --- workspace layout ---
    char* ws = (char*)d_ws; size_t o = 0;
    auto alloc = [&](size_t bytes) { char* p = ws + o; o = (o + bytes + 255) & ~(size_t)255; return p; };
    __bf16* wbf1_off  = (__bf16*)alloc((size_t)32  * 2304 * 2);
    __bf16* wbf1_main = (__bf16*)alloc((size_t)128 * 2304 * 2);
    __bf16* wbf2_off  = (__bf16*)alloc((size_t)32  * 1152 * 2);
    __bf16* wbf2_main = (__bf16*)alloc((size_t)256 * 1152 * 2);
    float*  om_raw = (float*)alloc((size_t)B * 27 * HW * 4);
    float*  om2    = (float*)alloc((size_t)B * 27 * HW * 4);
    float*  ph     = (float*)alloc((size_t)B * 27 * 80 * 4);
    float*  pw     = (float*)alloc((size_t)B * 27 * 80 * 4);
    float*  pch    = (float*)alloc((size_t)B * 27 * 4);
    float*  sh     = (float*)alloc((size_t)B * 27 * 80 * 4);
    float*  sw     = (float*)alloc((size_t)B * 27 * 80 * 4);
    float*  sc     = (float*)alloc((size_t)B * 27 * 4);
    float*  y1     = (float*)alloc((size_t)B * 128 * HW * 4);

    auto cvt = [&](const float* src, __bf16* dst, int rows, int K, int rowsPad) {
        int total = rowsPad * K;
        cvt_w_bf16<<<(total + 255) / 256, 256, 0, stream>>>(src, dst, rows, K, rowsPad);
    };
    cvt(OFFW1, wbf1_off,  27,  2304, 32);
    cvt(W1,    wbf1_main, 128, 2304, 128);
    cvt(OFFW2, wbf2_off,  27,  1152, 32);
    cvt(W2,    wbf2_main, 256, 1152, 256);

    const dim3 gtile(HW / 16, B);
    const int totOm = B * 27 * HW;

    // ---------------- dcn_mpca #1 (256 -> 128) ----------------
    offconv_wmma<256><<<gtile, 32, 0, stream>>>(X, wbf1_off, OFFB1, om_raw);
    mpca_pool<<<B * 27, 80, 0, stream>>>(om_raw, ph, pw, pch);
    mpca_mid<<<B, 256, 0, stream>>>(ph, pw, pch, M1, sh, sw, sc);
    mpca_apply<<<(totOm + 255) / 256, 256, 0, stream>>>(om_raw, sh, sw, sc, om2, totOm);
    deform_wmma<256, 8, false><<<gtile, 32, 0, stream>>>(
        X, om2, wbf1_main, B1, BN1g, BN1b, BN1m, BN1v, nullptr, y1);

    // ---------------- dcn_mpca #2 (128 -> 256) + residual ----------------
    offconv_wmma<128><<<gtile, 32, 0, stream>>>(y1, wbf2_off, OFFB2, om_raw);
    mpca_pool<<<B * 27, 80, 0, stream>>>(om_raw, ph, pw, pch);
    mpca_mid<<<B, 256, 0, stream>>>(ph, pw, pch, M2, sh, sw, sc);
    mpca_apply<<<(totOm + 255) / 256, 256, 0, stream>>>(om_raw, sh, sw, sc, om2, totOm);
    deform_wmma<128, 16, true><<<gtile, 32, 0, stream>>>(
        y1, om2, wbf2_main, B2, BN2g, BN2b, BN2m, BN2v, X, (float*)d_out);
}
